// MultiHeadAttentionV4_38508676776103
// MI455X (gfx1250) — compile-verified, matrix operand order checked by reading, not run on hardware
//
#include <hip/hip_runtime.h>

// ---------------------------------------------------------------------------
// GQA attention block (QKV proj + QK-RMSNorm + GQA softmax attn + xsa + out
// proj) for MI455X (gfx1250, wave32, WMMA). bf16 compute, fp32 accumulate.
// Round 3: Tensor Data Mover (TENSORcnt) staging of attention K tiles via
// the __builtin_amdgcn_tensor_load_to_lds descriptor DMA; async (ASYNCcnt)
// double-buffered staging kept for GEMM B tiles.
// ---------------------------------------------------------------------------

typedef __attribute__((ext_vector_type(16))) __bf16 v16bf;
typedef __attribute__((ext_vector_type(8)))  float  v8f;
typedef __attribute__((ext_vector_type(4)))  unsigned v4u;
typedef __attribute__((ext_vector_type(8)))  int      v8i;
typedef __attribute__((ext_vector_type(4)))  int      v4i;

#define DEVINL __device__ __forceinline__

#if defined(__has_builtin)
#  if __has_builtin(__builtin_amdgcn_tensor_load_to_lds) && \
      __has_builtin(__builtin_amdgcn_s_wait_tensorcnt)
#    define HAVE_TDM 1
#  endif
#endif
#ifndef HAVE_TDM
#  define HAVE_TDM 0
#endif

constexpr int EMBED   = 2048;
constexpr int HEADS   = 16;
constexpr int KVHEADS = 4;
constexpr int HD      = 128;   // head dim
constexpr int KVD     = 512;   // kv width
constexpr int BB      = 2;
constexpr int NN      = 2048;
constexpr int ROWS    = BB * NN;            // 4096
constexpr float RMS_EPS = 1.1920928955078125e-7f;   // FLT_EPSILON
constexpr float QSCALE  = 0.08838834764831845f;     // 128^-0.5

DEVINL __bf16 f2bf(float f) {
    unsigned u = __builtin_bit_cast(unsigned, f);
    u += 0x7fffu + ((u >> 16) & 1u);                // round-to-nearest-even
    unsigned short h = (unsigned short)(u >> 16);
    return __builtin_bit_cast(__bf16, h);
}
DEVINL float bf2f(__bf16 b) {
    unsigned short h = __builtin_bit_cast(unsigned short, b);
    unsigned u = ((unsigned)h) << 16;
    return __builtin_bit_cast(float, u);
}

// --- async global->LDS helpers (gfx1250; tracked with ASYNCcnt) ------------
DEVINL unsigned lds_off_of(void* p) {
    // addrspace(3) pointers are byte offsets within the wave's LDS window
    return (unsigned)(size_t)(__attribute__((address_space(3))) char*)p;
}
DEVINL void async_copy_b128(void* lds_dst, const void* gsrc) {
    unsigned l = lds_off_of(lds_dst);
    asm volatile("global_load_async_to_lds_b128 %0, %1, off"
                 :
                 : "v"(l), "v"(gsrc)
                 : "memory");
}
DEVINL void wait_async0() {
    asm volatile("s_wait_asynccnt 0x0" ::: "memory");
}

// --- Tensor Data Mover: 2D tile (16-bit elems) global -> LDS ---------------
// D# packing per CDNA5 ISA ch.8 (group0: count/lds/global/type,
// group1: data_size + dims + tile dims + dim0 stride). Issue once per wave;
// completion tracked with TENSORcnt.
#if HAVE_TDM
DEVINL void tdm_load_tile_2d_b16(unsigned lds_byte_addr, const void* gptr,
                                 unsigned tile_d0, unsigned tile_d1,
                                 unsigned d0_stride_elems) {
    const unsigned long long ga = (unsigned long long)(size_t)gptr;
    v4u g0 = { 1u,                                  // count=1, user descriptor
               lds_byte_addr,                       // lds_addr[31:0]
               (unsigned)ga,                        // global_addr[31:0]
               (unsigned)(ga >> 32) | (2u << 30) }; // global_addr[56:32]|type=2
    v8i g1 = { (int)(1u << 16),                          // data_size=1 (2B)
               (int)((tile_d0 & 0xffffu) << 16),         // tensor_dim0[15:0]
               (int)((tile_d1 & 0xffffu) << 16),         // tensor_dim1[15:0]
               (int)((tile_d0 & 0xffffu) << 16),         // tile_dim0
               (int)(tile_d1 & 0xffffu),                 // tile_dim1
               (int)d0_stride_elems,                     // dim0_stride[31:0]
               0, 0 };
    v4i gz = {0, 0, 0, 0};
#if __clang_major__ >= 23
    v8i gz8 = {0, 0, 0, 0, 0, 0, 0, 0};
    __builtin_amdgcn_tensor_load_to_lds(g0, g1, gz, gz, gz8, 0);
#else
    __builtin_amdgcn_tensor_load_to_lds(g0, g1, gz, gz, 0);
#endif
}
#endif

// ---------------------------------------------------------------------------
// fp32 -> bf16 pack (grid-stride)
// ---------------------------------------------------------------------------
__global__ void cvt_f32_bf16(const float* __restrict__ s,
                             __bf16* __restrict__ d, int n) {
    int i = blockIdx.x * blockDim.x + threadIdx.x;
    int stride = gridDim.x * blockDim.x;
    for (; i < n; i += stride) d[i] = f2bf(s[i]);
}

// ---------------------------------------------------------------------------
// bf16 GEMM with Linear semantics: C[m,n] = sum_k A[m,k] * W[n,k]
//   A: [M x K] bf16 row-major, W: [N x K] bf16 row-major.
// Wave tile: 16M x 64N.  Block: 8 waves stacked along M -> 128M x 64N.
// W tile (64 cols x 32 k = 4KB) staged to LDS with async b128 copies,
// double-buffered: async DMA of tile k+1 overlaps WMMA on tile k.
// ---------------------------------------------------------------------------
template <bool OUT_BF16>
__global__ __launch_bounds__(256)
void gemm_xWT(const __bf16* __restrict__ A, int lda,
              const __bf16* __restrict__ W, int ldw,
              void* __restrict__ Cv, int ldc, int K) {
    __shared__ __align__(128) __bf16 Wt[2][64][32];   // [buf][col][k]

    const int tid  = threadIdx.x;
    const int lane = tid & 31;
    const int wave = tid >> 5;
    const int m0   = blockIdx.y * 128 + wave * 16;
    const int nb   = blockIdx.x * 64;
    const int l16  = lane & 15;
    const int kof  = (lane >> 4) << 4;

    const __bf16* ap = A + (size_t)(m0 + l16) * lda + kof;

    // per-thread stage assignment: one b128 (8 bf16) per thread per k-step
    const int scol = tid >> 2;          // 0..63: W row (output col)
    const int sch  = (tid & 3) << 3;    // 0,8,16,24: k chunk
    const __bf16* wsrc = W + (size_t)(nb + scol) * ldw + sch;

    v8f acc0 = {0,0,0,0,0,0,0,0};
    v8f acc1 = acc0, acc2 = acc0, acc3 = acc0;

    // prologue: stage first W tile
    async_copy_b128(&Wt[0][scol][sch], wsrc);
    wait_async0();
    __syncthreads();

    int bufi = 0;
    for (int k0 = 0; k0 < K; k0 += 32) {
        // kick off async DMA of the next tile into the other buffer
        if (k0 + 32 < K)
            async_copy_b128(&Wt[bufi ^ 1][scol][sch], wsrc + k0 + 32);

        v16bf a  = *(const v16bf*)(ap + k0);
        v16bf b0 = *(const v16bf*)&Wt[bufi][l16][kof];
        v16bf b1 = *(const v16bf*)&Wt[bufi][16 + l16][kof];
        v16bf b2 = *(const v16bf*)&Wt[bufi][32 + l16][kof];
        v16bf b3 = *(const v16bf*)&Wt[bufi][48 + l16][kof];
        acc0 = __builtin_amdgcn_wmma_f32_16x16x32_bf16(false, a, false, b0, (short)0, acc0, false, false);
        acc1 = __builtin_amdgcn_wmma_f32_16x16x32_bf16(false, a, false, b1, (short)0, acc1, false, false);
        acc2 = __builtin_amdgcn_wmma_f32_16x16x32_bf16(false, a, false, b2, (short)0, acc2, false, false);
        acc3 = __builtin_amdgcn_wmma_f32_16x16x32_bf16(false, a, false, b3, (short)0, acc3, false, false);
        // speculative prefetch of upcoming A chunk (-> global_prefetch_b8)
        __builtin_prefetch(ap + k0 + 512, 0, 3);

        wait_async0();       // next tile's DMA has landed
        __syncthreads();     // and everyone is done reading the current one
        bufi ^= 1;
    }

    // C layout: VGPR r holds M=r (lanes 0-15) / M=r+8 (lanes 16-31), N=lane&15
    const int mrow = m0 + 8 * (lane >> 4);
    const int ncol = nb + l16;
#pragma unroll
    for (int r = 0; r < 8; r++) {
        size_t base = (size_t)(mrow + r) * ldc + ncol;
        if constexpr (OUT_BF16) {
            __bf16* C = (__bf16*)Cv;
            C[base]      = f2bf(acc0[r]);
            C[base + 16] = f2bf(acc1[r]);
            C[base + 32] = f2bf(acc2[r]);
            C[base + 48] = f2bf(acc3[r]);
        } else {
            float* C = (float*)Cv;
            C[base]      = acc0[r];
            C[base + 16] = acc1[r];
            C[base + 32] = acc2[r];
            C[base + 48] = acc3[r];
        }
    }
}

// ---------------------------------------------------------------------------
// RMS norm over rows of 128 (flat view [nrows,128]); one wave per row.
// Folds an output scale (attention 1/sqrt(HD) for Q). fp32 in -> bf16 out.
// ---------------------------------------------------------------------------
__global__ __launch_bounds__(256)
void rmsnorm_rows(const float* __restrict__ src, __bf16* __restrict__ dst,
                  int nrows, float outscale) {
    const int lane = threadIdx.x & 31;
    const int wave = threadIdx.x >> 5;
    const int row  = blockIdx.x * 8 + wave;
    if (row >= nrows) return;

    const float4 v = *(const float4*)(src + (size_t)row * 128 + lane * 4);
    float ss = v.x * v.x + v.y * v.y + v.z * v.z + v.w * v.w;
#pragma unroll
    for (int off = 16; off; off >>= 1) ss += __shfl_xor(ss, off, 32);
    const float r = rsqrtf(ss * (1.f / 128.f) + RMS_EPS) * outscale;

    __bf16* o = dst + (size_t)row * 128 + lane * 4;
    o[0] = f2bf(v.x * r);
    o[1] = f2bf(v.y * r);
    o[2] = f2bf(v.z * r);
    o[3] = f2bf(v.w * r);
}

// ---------------------------------------------------------------------------
// Flash-style GQA attention + xsa correction.
// Grid: (N/128, HEADS, B).  Block: 256 threads = 8 waves; wave -> 16 q rows.
// Per 32-key tile: K tile DMA'd to LDS by the Tensor Data Mover (wave 0
// issues one 2D-tile descriptor; completion via s_wait_tensorcnt), with an
// async-copy fallback when the TDM builtin is unavailable. V tile staged
// TRANSPOSED (VGPR scatter) so the PV B-fragment is contiguous along the
// key (contraction) dimension.
// ---------------------------------------------------------------------------
__global__ __launch_bounds__(256)
void attention_xsa(const __bf16* __restrict__ Qb, const __bf16* __restrict__ Kb,
                   const __bf16* __restrict__ Vb, __bf16* __restrict__ Y) {
    __shared__ __align__(128) __bf16 Kt[32][128];     // [key][d]
    __shared__ __align__(128) __bf16 Vt[128][32];     // [d][key] (transposed)
    __shared__ __align__(128) __bf16 Pb[8][16][32];   // per-wave P transpose

    const int tid  = threadIdx.x;
    const int lane = tid & 31;
    const int wave = tid >> 5;
    const int q0   = blockIdx.x * 128;
    const int h    = blockIdx.y;
    const int b    = blockIdx.z;
    const int kvh  = h >> 2;                 // GQA: 4 q-heads per kv-head
    const int l16  = lane & 15;
    const int kof  = (lane >> 4) << 4;

    // Q fragments for this wave's 16 rows: reused across the whole key loop
    const size_t qrow = (size_t)(b * NN + q0 + wave * 16 + l16);
    const __bf16* qp  = Qb + qrow * EMBED + h * HD + kof;
    v16bf qf0 = *(const v16bf*)(qp + 0);
    v16bf qf1 = *(const v16bf*)(qp + 32);
    v16bf qf2 = *(const v16bf*)(qp + 64);
    v16bf qf3 = *(const v16bf*)(qp + 96);

    v8f acc[8];
#pragma unroll
    for (int j = 0; j < 8; j++) acc[j] = (v8f){0,0,0,0,0,0,0,0};
    float mrow[8], lrow[8];
#pragma unroll
    for (int r = 0; r < 8; r++) { mrow[r] = -3.0e38f; lrow[r] = 0.f; }

    const int srow = tid >> 3;         // 0..31: staged key row
    const int scol = (tid & 7) << 4;   // 0,16,...,112: d chunk
#if HAVE_TDM
    const unsigned ldsKt = lds_off_of(&Kt[0][0]);
#endif

    for (int key0 = 0; key0 < NN; key0 += 32) {
        // ---- stage K tile (TDM descriptor DMA); V tile via VGPR scatter
        {
            const size_t kvrow =
                (size_t)(b * NN + key0 + srow) * KVD + kvh * HD + scol;
#if HAVE_TDM
            if (wave == 0) {
                const __bf16* ktile =
                    Kb + (size_t)(b * NN + key0) * KVD + kvh * HD;
                tdm_load_tile_2d_b16(ldsKt, ktile, /*d0=*/HD, /*d1=*/32,
                                     /*stride=*/KVD);
            }
#else
            const __bf16* ksrc = Kb + kvrow;
            async_copy_b128(&Kt[srow][scol], ksrc);
            async_copy_b128(&Kt[srow][scol + 8], ksrc + 8);
#endif
            v16bf vv = *(const v16bf*)(Vb + kvrow);
#pragma unroll
            for (int i = 0; i < 16; i++) Vt[scol + i][srow] = vv[i];
        }
#if HAVE_TDM
        if (wave == 0) __builtin_amdgcn_s_wait_tensorcnt(0);
#else
        wait_async0();
#endif
        __syncthreads();

        // ---- S[16q x 32k] = Q . K^T  (two 16-key halves, K-dim = 128)
        v8f s0 = {0,0,0,0,0,0,0,0};
        v8f s1 = s0;
#pragma unroll
        for (int d0 = 0; d0 < 4; d0++) {
            v16bf qa = (d0 == 0) ? qf0 : (d0 == 1) ? qf1 : (d0 == 2) ? qf2 : qf3;
            v16bf kb0 = *(const v16bf*)&Kt[l16][d0 * 32 + kof];
            v16bf kb1 = *(const v16bf*)&Kt[16 + l16][d0 * 32 + kof];
            s0 = __builtin_amdgcn_wmma_f32_16x16x32_bf16(false, qa, false, kb0, (short)0, s0, false, false);
            s1 = __builtin_amdgcn_wmma_f32_16x16x32_bf16(false, qa, false, kb1, (short)0, s1, false, false);
        }

        // ---- online softmax (rows live in {r, lane-half}; cols in lanes)
        float alpha[8];
#pragma unroll
        for (int r = 0; r < 8; r++) {
            float mx = fmaxf(s0[r], s1[r]);
#pragma unroll
            for (int off = 8; off; off >>= 1) mx = fmaxf(mx, __shfl_xor(mx, off, 32));
            const float mn = fmaxf(mrow[r], mx);
            const float al = __expf(mrow[r] - mn);
            const float p0 = __expf(s0[r] - mn);
            const float p1 = __expf(s1[r] - mn);
            float rs = p0 + p1;
#pragma unroll
            for (int off = 8; off; off >>= 1) rs += __shfl_xor(rs, off, 32);
            lrow[r]  = lrow[r] * al + rs;
            mrow[r]  = mn;
            alpha[r] = al;
            s0[r] = p0;
            s1[r] = p1;
        }
#pragma unroll
        for (int j = 0; j < 8; j++)
#pragma unroll
            for (int r = 0; r < 8; r++) acc[j][r] = acc[j][r] * alpha[r];

        // ---- P (C-layout) -> A-fragment via per-wave LDS transpose
#pragma unroll
        for (int r = 0; r < 8; r++) {
            const int m = r + 8 * (lane >> 4);
            Pb[wave][m][l16]      = f2bf(s0[r]);
            Pb[wave][m][16 + l16] = f2bf(s1[r]);
        }
        asm volatile("" ::: "memory");   // per-wave DS ops are in-order
        v16bf pf = *(const v16bf*)&Pb[wave][l16][kof];

        // ---- acc[16q x 128d] += P . V   (contraction over 32 keys)
#pragma unroll
        for (int j = 0; j < 8; j++) {
            v16bf vb = *(const v16bf*)&Vt[j * 16 + l16][kof];
            acc[j] = __builtin_amdgcn_wmma_f32_16x16x32_bf16(false, pf, false, vb, (short)0, acc[j], false, false);
        }
        __syncthreads();
    }

    // ---- normalize by softmax denominator
#pragma unroll
    for (int r = 0; r < 8; r++) {
        const float linv = 1.f / lrow[r];
#pragma unroll
        for (int j = 0; j < 8; j++) acc[j][r] = acc[j][r] * linv;
    }

    // ---- xsa: y -= (y . v_hat) v_hat with v at the same (query) position
#pragma unroll
    for (int r = 0; r < 8; r++) {
        const int m = r + 8 * (lane >> 4);
        const size_t row = (size_t)(b * NN + q0 + wave * 16 + m);
        const __bf16* vp = Vb + row * KVD + kvh * HD + l16;
        float nv[8];
        float n2 = 0.f;
#pragma unroll
        for (int j = 0; j < 8; j++) {
            nv[j] = bf2f(vp[j * 16]);
            n2 += nv[j] * nv[j];
        }
#pragma unroll
        for (int off = 8; off; off >>= 1) n2 += __shfl_xor(n2, off, 32);
        const float inv = 1.f / fmaxf(sqrtf(n2), 1e-12f);
        float d1 = 0.f;
#pragma unroll
        for (int j = 0; j < 8; j++) d1 += acc[j][r] * nv[j];
#pragma unroll
        for (int off = 8; off; off >>= 1) d1 += __shfl_xor(d1, off, 32);
        const float dd = d1 * inv * inv;

        __bf16* yp = Y + row * EMBED + h * HD + l16;
#pragma unroll
        for (int j = 0; j < 8; j++) yp[j * 16] = f2bf(acc[j][r] - dd * nv[j]);
    }
}

// ---------------------------------------------------------------------------
// host-side launcher
// ---------------------------------------------------------------------------
extern "C" void kernel_launch(void* const* d_in, const int* in_sizes, int n_in,
                              void* d_out, int out_size, void* d_ws,
                              size_t ws_size, hipStream_t stream) {
    (void)in_sizes; (void)n_in; (void)out_size; (void)ws_size;
    const float* x  = (const float*)d_in[0];
    const float* Wq = (const float*)d_in[1];
    const float* Wk = (const float*)d_in[2];
    const float* Wv = (const float*)d_in[3];
    const float* Wo = (const float*)d_in[4];
    float* out = (float*)d_out;

    char* ws = (char*)d_ws;
    size_t off = 0;
    auto carve = [&](size_t bytes) -> void* {
        void* p = ws + off;
        off += (bytes + 255) & ~(size_t)255;
        return p;
    };

    __bf16* xb  = (__bf16*)carve((size_t)ROWS * EMBED * 2);
    __bf16* wqb = (__bf16*)carve((size_t)EMBED * EMBED * 2);
    __bf16* wkb = (__bf16*)carve((size_t)KVD * EMBED * 2);
    __bf16* wvb = (__bf16*)carve((size_t)KVD * EMBED * 2);
    __bf16* wob = (__bf16*)carve((size_t)EMBED * EMBED * 2);
    float*  Qf  = (float*)carve((size_t)ROWS * EMBED * 4);
    float*  Kf  = (float*)carve((size_t)ROWS * KVD * 4);
    __bf16* Qbn = (__bf16*)carve((size_t)ROWS * EMBED * 2);
    __bf16* Kbn = (__bf16*)carve((size_t)ROWS * KVD * 2);
    __bf16* Vbb = (__bf16*)carve((size_t)ROWS * KVD * 2);
    __bf16* Yb  = (__bf16*)carve((size_t)ROWS * EMBED * 2);

    // 1) pack fp32 -> bf16
    cvt_f32_bf16<<<2048, 256, 0, stream>>>(x,  xb,  ROWS * EMBED);
    cvt_f32_bf16<<<1024, 256, 0, stream>>>(Wq, wqb, EMBED * EMBED);
    cvt_f32_bf16<<<512,  256, 0, stream>>>(Wk, wkb, KVD * EMBED);
    cvt_f32_bf16<<<512,  256, 0, stream>>>(Wv, wvb, KVD * EMBED);
    cvt_f32_bf16<<<1024, 256, 0, stream>>>(Wo, wob, EMBED * EMBED);

    // 2) projections: y = x @ W^T
    gemm_xWT<false><<<dim3(EMBED / 64, ROWS / 128), 256, 0, stream>>>(
        xb, EMBED, wqb, EMBED, Qf, EMBED, EMBED);
    gemm_xWT<false><<<dim3(KVD / 64, ROWS / 128), 256, 0, stream>>>(
        xb, EMBED, wkb, EMBED, Kf, KVD, EMBED);
    gemm_xWT<true><<<dim3(KVD / 64, ROWS / 128), 256, 0, stream>>>(
        xb, EMBED, wvb, EMBED, Vbb, KVD, EMBED);

    // 3) QK RMSNorm (per 128-wide head row); fold 1/sqrt(HD) into Q
    rmsnorm_rows<<<ROWS * HEADS / 8, 256, 0, stream>>>(Qf, Qbn, ROWS * HEADS, QSCALE);
    rmsnorm_rows<<<ROWS * KVHEADS / 8, 256, 0, stream>>>(Kf, Kbn, ROWS * KVHEADS, 1.0f);

    // 4) attention + xsa
    attention_xsa<<<dim3(NN / 128, HEADS, BB), 256, 0, stream>>>(Qbn, Kbn, Vbb, Yb);

    // 5) output projection -> fp32 d_out
    gemm_xWT<false><<<dim3(EMBED / 64, ROWS / 128), 256, 0, stream>>>(
        Yb, EMBED, wob, EMBED, out, EMBED, EMBED);
}
